// Combine_Stack_with_attn_41566693491391
// MI455X (gfx1250) — matrix operand order repaired and checked by reading, not compiled, not run on hardware
//
#include <hip/hip_runtime.h>

typedef __attribute__((ext_vector_type(16))) _Float16 v16h;
typedef __attribute__((ext_vector_type(8)))  float    v8f;

#define MODEL 512
#define OUTF  256
#define WSZ   16
#define WIN   33
#define TLEN  2048

// ---------------------------------------------------------------------------
// Pack an f32 KxN weight matrix into WMMA B-fragment lane order (f16).
// B tile = 32x16 (K x N). Per 512-element tile: lane (0..31) holds 16 halves:
//   element e -> K = kt*32 + (lane/16)*16 + e,  N = nt*16 + (lane&15)
// Stored so each lane reads its 16 halves contiguously (2 x b128 loads).
// ---------------------------------------------------------------------------
__global__ void pack_weight(const float* __restrict__ W, _Float16* __restrict__ Wp,
                            int Kdim, int Ndim) {
    int h = blockIdx.x * blockDim.x + threadIdx.x;
    int total = Kdim * Ndim;
    if (h >= total) return;
    int tile   = h >> 9;         // / 512
    int within = h & 511;
    int lane   = within >> 4;
    int e      = within & 15;
    int ntiles = Ndim >> 4;
    int kt = tile / ntiles;
    int nt = tile % ntiles;
    int K   = kt * 32 + ((lane >> 4) << 4) + e;
    int col = nt * 16 + (lane & 15);
    Wp[h] = (_Float16)W[(size_t)K * Ndim + col];
}

__global__ void f32_to_f16(const float* __restrict__ s, _Float16* __restrict__ d, int n) {
    int i = blockIdx.x * blockDim.x + threadIdx.x;
    if (i < n) d[i] = (_Float16)s[i];
}

// ---------------------------------------------------------------------------
// WMMA GEMM: C(M x 256) = A(M x Kdim, f16 row-major) @ Bp (packed f16).
// Block = 128 threads (4 waves), blockIdx.x = 32-row strip (2 M-tiles).
// Wave w covers columns [w*64, w*64+64) = 4 N-tiles for both M-tiles.
// Per K-tile: 2 A-frag loads + 4 B-frag loads feed 8 WMMAs (each B frag is
// reused by 2 independent accumulator chains). MPB=2 keeps the register
// footprint (~8 acc frags = 64 VGPRs + 2 A frags + B frag) well under the
// 256-VGPR addressable budget -> no scratch spills (MPB=4 spilled).
// fuse==1: C = sigmoid(acc + addC + bias)  (final layer epilogue).
// ---------------------------------------------------------------------------
#define MPB 2   // M-tiles per block

__global__ void gemm_wmma(const _Float16* __restrict__ A,
                          const _Float16* __restrict__ Bp,
                          const float* __restrict__ addC,
                          const float* __restrict__ bias,
                          float* __restrict__ C,
                          int Kdim, int fuse) {
    const int lane  = threadIdx.x & 31;
    const int wave  = threadIdx.x >> 5;       // 0..3
    const int mt0   = blockIdx.x * MPB;       // first M-tile
    const int lhalf = lane >> 4;

    v8f acc[MPB][4] = {};

    const int ntiles = OUTF / 16;             // 16
    const int ktc    = Kdim >> 5;

    for (int kt = 0; kt < ktc; ++kt) {
        // A fragments: contiguous k order (k = 16*(lane/16) + e), matching packed B
        union { v16h v; unsigned u[8]; } af[MPB];
#pragma unroll
        for (int m = 0; m < MPB; ++m) {
            const int mrow = (mt0 + m) * 16 + (lane & 15);
            const unsigned* arow =
                (const unsigned*)(A + (size_t)mrow * Kdim + (size_t)kt * 32);
#pragma unroll
            for (int p = 0; p < 8; ++p) {
                int k0 = 2 * p + 16 * lhalf;
                af[m].u[p] = arow[k0 >> 1];
            }
        }
#pragma unroll
        for (int j = 0; j < 4; ++j) {
            int nt = wave * 4 + j;
            union { v16h v; uint4 q[2]; } bf;
            const uint4* bp =
                (const uint4*)(Bp + ((size_t)(kt * ntiles + nt) * 32 + lane) * 16);
            bf.q[0] = bp[0];
            bf.q[1] = bp[1];
#pragma unroll
            for (int m = 0; m < MPB; ++m) {
                acc[m][j] = __builtin_amdgcn_wmma_f32_16x16x32_f16(
                    false, af[m].v, false, bf.v, (short)0, acc[m][j], false, false);
            }
        }
    }

#pragma unroll
    for (int m = 0; m < MPB; ++m) {
        const int rbase = (mt0 + m) * 16 + 8 * lhalf;  // VGPR r -> M = r + 8*(lane/16)
#pragma unroll
        for (int j = 0; j < 4; ++j) {
            int col = (wave * 4 + j) * 16 + (lane & 15);
#pragma unroll
            for (int r = 0; r < 8; ++r) {
                size_t idx = (size_t)(rbase + r) * OUTF + col;
                float z = acc[m][j][r];
                if (fuse) {
                    z += addC[idx] + bias[col];
                    z = 1.0f / (1.0f + __expf(-z));
                }
                C[idx] = z;
            }
        }
    }
}

// ---------------------------------------------------------------------------
// Per-(b,t) attention: scores over 33 window slots, softmax, weighted sum.
// Block = 256 threads (8 waves). Wave -> one window slot per pass (5 passes),
// 32 lanes x 8 channels, wave shuffle reduction. Softmax done serially by
// tid 0 (33 elements). Weighted sum: thread f over 33 slots.
// Out-of-range slots: P-term is 0 (zero padding) but Q + b_attn still apply.
// ---------------------------------------------------------------------------
__global__ void attn_kernel(const float* __restrict__ Q,
                            const float* __restrict__ P,
                            const float* __restrict__ enc,
                            const float* __restrict__ b_attn,
                            const float* __restrict__ v,
                            _Float16* __restrict__ weightedh,
                            float* __restrict__ a_out,
                            int T) {
    const int row = blockIdx.x;               // b*T + t
    const int t = row % T;
    const int rowbase = row - t;              // b*T
    const int tid = threadIdx.x;
    const int lane = tid & 31, wave = tid >> 5;

    __shared__ float sc[WIN];
    __shared__ float a_sh[WIN];

    const float* q = Q + (size_t)row * OUTF;

    for (int pass = 0; pass < 5; ++pass) {
        int w = pass * 8 + wave;
        if (w < WIN) {
            int tt = t + w - WSZ;
            bool inr = (tt >= 0) && (tt < T);
            float partial = 0.f;
#pragma unroll
            for (int i = 0; i < 8; ++i) {
                int o = lane + 32 * i;
                float pv = 0.f;
                if (inr) pv = P[(size_t)(rowbase + tt) * OUTF + o];
                float e = q[o] + pv + b_attn[o];
                partial += v[o] * tanhf(e);
            }
#pragma unroll
            for (int off = 16; off; off >>= 1)
                partial += __shfl_xor(partial, off, 32);
            if (lane == 0) sc[w] = partial;
        }
    }
    __syncthreads();

    if (tid == 0) {
        float mx = sc[0];
        for (int w = 1; w < WIN; ++w) mx = fmaxf(mx, sc[w]);
        float s = 0.f;
        for (int w = 0; w < WIN; ++w) { float e = __expf(sc[w] - mx); a_sh[w] = e; s += e; }
        float inv = 1.f / s;
        for (int w = 0; w < WIN; ++w) a_sh[w] *= inv;
    }
    __syncthreads();

    {   // weighted[f] = sum_w a_w * enc[t+w-16, f] (f32 source for accuracy)
        int f = tid;
        float acc = 0.f;
        for (int w = 0; w < WIN; ++w) {
            int tt = t + w - WSZ;
            if (tt >= 0 && tt < T)
                acc += a_sh[w] * enc[(size_t)(rowbase + tt) * OUTF + f];
        }
        weightedh[(size_t)row * OUTF + f] = (_Float16)acc;
    }
    if (tid < WIN) a_out[(size_t)row * WIN + tid] = a_sh[tid];
}

// ---------------------------------------------------------------------------
extern "C" void kernel_launch(void* const* d_in, const int* in_sizes, int n_in,
                              void* d_out, int out_size, void* d_ws, size_t ws_size,
                              hipStream_t stream) {
    (void)n_in; (void)out_size; (void)ws_size;
    const float* x      = (const float*)d_in[0];
    const float* enc    = (const float*)d_in[1];
    const float* W_attn = (const float*)d_in[2];
    const float* b_attn = (const float*)d_in[3];
    const float* v      = (const float*)d_in[4];
    const float* W_lin  = (const float*)d_in[5];
    const float* b_lin  = (const float*)d_in[6];

    const int Mtot = in_sizes[0] / MODEL;     // B*T = 8192
    const int T = TLEN;

    char* ws = (char*)d_ws;
    size_t off = 0;
    auto alloc = [&](size_t bytes) {
        char* p = ws + off;
        off += (bytes + 255) & ~(size_t)255;
        return p;
    };
    _Float16* WhP  = (_Float16*)alloc((size_t)MODEL * OUTF * 2);
    _Float16* WeP  = (_Float16*)alloc((size_t)OUTF  * OUTF * 2);
    _Float16* Wl1P = (_Float16*)alloc((size_t)MODEL * OUTF * 2);
    _Float16* Wl2P = (_Float16*)alloc((size_t)OUTF  * OUTF * 2);
    _Float16* xh   = (_Float16*)alloc((size_t)Mtot * MODEL * 2);
    _Float16* ench = (_Float16*)alloc((size_t)Mtot * OUTF * 2);
    float*    Qb   = (float*)   alloc((size_t)Mtot * OUTF * 4);
    float*    Pb   = (float*)   alloc((size_t)Mtot * OUTF * 4);
    float*    Xl   = (float*)   alloc((size_t)Mtot * OUTF * 4);
    _Float16* wgh  = (_Float16*)alloc((size_t)Mtot * OUTF * 2);

    float* out_main = (float*)d_out;                       // (B,T,256)
    float* out_a    = out_main + (size_t)Mtot * OUTF;      // (B,T,1,33)

    // Pack weights into WMMA B-fragment order (f16)
    {
        int n1 = MODEL * OUTF, n2 = OUTF * OUTF;
        pack_weight<<<(n1 + 255) / 256, 256, 0, stream>>>(W_attn, WhP, MODEL, OUTF);
        pack_weight<<<(n2 + 255) / 256, 256, 0, stream>>>(W_attn + (size_t)MODEL * OUTF, WeP, OUTF, OUTF);
        pack_weight<<<(n1 + 255) / 256, 256, 0, stream>>>(W_lin, Wl1P, MODEL, OUTF);
        pack_weight<<<(n2 + 255) / 256, 256, 0, stream>>>(W_lin + (size_t)MODEL * OUTF, Wl2P, OUTF, OUTF);
    }
    // Convert activations to f16
    {
        int nx = Mtot * MODEL, ne = Mtot * OUTF;
        f32_to_f16<<<(nx + 255) / 256, 256, 0, stream>>>(x, xh, nx);
        f32_to_f16<<<(ne + 255) / 256, 256, 0, stream>>>(enc, ench, ne);
    }
    // Q = x@Wh ; P = enc@We ; Xl = x@Wlin[:512]
    gemm_wmma<<<Mtot / (16 * MPB), 128, 0, stream>>>(xh,   WhP,  nullptr, nullptr, Qb, MODEL, 0);
    gemm_wmma<<<Mtot / (16 * MPB), 128, 0, stream>>>(ench, WeP,  nullptr, nullptr, Pb, OUTF,  0);
    gemm_wmma<<<Mtot / (16 * MPB), 128, 0, stream>>>(xh,   Wl1P, nullptr, nullptr, Xl, MODEL, 0);
    // tanh-attention: scores, softmax, weighted
    attn_kernel<<<Mtot, 256, 0, stream>>>(Qb, Pb, enc, b_attn, v, wgh, out_a, T);
    // out = sigmoid(Xl + weighted@Wlin[512:] + b_lin)
    gemm_wmma<<<Mtot / (16 * MPB), 128, 0, stream>>>(wgh, Wl2P, Xl, b_lin, out_main, OUTF, 1);
}
